// LSTMAutoEncoder_20426864460005
// MI455X (gfx1250) — matrix operand order repaired and checked by reading, not compiled, not run on hardware
//
#include <hip/hip_runtime.h>
#include <hip/hip_bf16.h>

// ---------------------------------------------------------------------------
// LSTM autoencoder for MI455X (gfx1250): persistent per-batch-tile workgroups,
// fused [x_t | h] GEMM via v_wmma_f32_16x16x32_bf16, f32 accumulate.
// Weights stream from L2 every timestep; an opaque-pointer asm barrier stops
// LICM from hoisting the (t-invariant) weight loads into spilled registers.
// ---------------------------------------------------------------------------

typedef __bf16 v16bf __attribute__((ext_vector_type(16)));
typedef float  v8f   __attribute__((ext_vector_type(8)));

struct Frag32 { uint4 lo; uint4 hi; };   // 32 bytes = 16 bf16 = one WMMA A/B operand

__device__ __forceinline__ unsigned short f2bf(float f) {
    unsigned int u = __float_as_uint(f);
    u += 0x7fffu + ((u >> 16) & 1u);     // round-to-nearest-even
    return (unsigned short)(u >> 16);
}
__device__ __forceinline__ float fsigmoid(float x) {
    return 1.0f / (1.0f + __expf(-x));
}
__device__ __forceinline__ float ftanh_fast(float x) {
    return 2.0f / (1.0f + __expf(-2.0f * x)) - 1.0f;
}

// ---------------------------------------------------------------------------
// Weight repack: W_cat = [Wih | Whh] (4H x K, K = Din+H) into per-fragment,
// per-lane-contiguous bf16 order:
//   packed[(((ntile*KT + kt)*32 + lane)*16 + e)] = W_cat[n, k]
//   n = ntile*16 + (lane&15),  k = kt*32 + (lane>>4)*16 + e
// so each lane's 16 bf16 B-fragment elements are one 32-byte contiguous load.
// ---------------------------------------------------------------------------
__global__ void pack_weights(const float* __restrict__ Wih,
                             const float* __restrict__ Whh,
                             unsigned short* __restrict__ out,
                             int Din, int Hid) {
    const int  K  = Din + Hid;
    const int  KT = K / 32;
    const long total = (long)(4 * Hid) * K;
    long idx = (long)blockIdx.x * blockDim.x + threadIdx.x;
    if (idx >= total) return;
    int  e     = (int)(idx & 15);
    int  lane  = (int)((idx >> 4) & 31);
    long rest  = idx >> 9;
    int  kt    = (int)(rest % KT);
    int  ntile = (int)(rest / KT);
    int  n = ntile * 16 + (lane & 15);
    int  k = kt * 32 + (lane >> 4) * 16 + e;
    float v = (k < Din) ? Wih[(long)n * Din + k] : Whh[(long)n * Hid + (k - Din)];
    out[idx] = f2bf(v);
}

__global__ void pack_bias(const float* __restrict__ bih,
                          const float* __restrict__ bhh,
                          float* __restrict__ out, int n4h) {
    int i = blockIdx.x * blockDim.x + threadIdx.x;
    if (i < n4h) out[i] = bih[i] + bhh[i];
}

// ---------------------------------------------------------------------------
// Persistent LSTM layer kernel. Grid = B/16 workgroups (16 batch rows each),
// one wave per 16-wide hidden-column tile (H/16 waves). Loops over T steps:
//   gates[16,4H] = [x_t | h] @ Wcat^T   (WMMA bf16 -> f32)
//   c = sig(f)*c + sig(i)*tanh(g);  h = sig(o)*tanh(c)
// h lives in LDS (bf16, appended after the x columns so A-fragments read the
// concatenated K dimension directly); c lives in accumulator-layout VGPRs.
// ---------------------------------------------------------------------------
enum { IN_SEQ_F32 = 0, IN_SEQ_BF16 = 1, IN_BCAST_F32 = 2 };
enum { OUT_SEQ_BF16 = 0, OUT_LAST_F32 = 1, OUT_SEQ_F32 = 2 };

template <int DIN, int HID, int INMODE, int OUTMODE>
__global__ __launch_bounds__(32 * (HID / 16))
void lstm_layer(const void* __restrict__ in_seq,
                const unsigned short* __restrict__ wpack,
                const float* __restrict__ bb,
                void* __restrict__ out,
                int T_) {
    constexpr int K    = DIN + HID;
    constexpr int KT   = K / 32;
    constexpr int H16  = HID / 16;
    constexpr int LDST = K + 8;               // bf16 row stride (+16B pad: bank-spread)
    __shared__ __align__(16) unsigned short smem[16 * LDST];

    const int tid      = threadIdx.x;
    const int lane     = tid & 31;
    const int wv       = tid >> 5;            // hidden-column tile owned by this wave
    const int bg       = blockIdx.x;          // 16-row batch group
    const int nthreads = 32 * H16;
    const int m        = lane & 15;           // A row / C column selector
    const int hh       = lane >> 4;           // K-half / C row-half selector
    const int col      = wv * 16 + m;         // hidden column for this lane's C column

    for (int i = tid; i < 16 * LDST; i += nthreads) smem[i] = 0;

    const float bi  = bb[0 * HID + col];
    const float bf  = bb[1 * HID + col];
    const float bg_ = bb[2 * HID + col];
    const float bo  = bb[3 * HID + col];

    if (INMODE == IN_BCAST_F32) {             // decoder L0: repeated hT, load once
        const float* src = (const float*)in_seq;   // [B, DIN]
        for (int i = tid; i < 16 * DIN; i += nthreads) {
            int r = i / DIN, c = i % DIN;
            smem[r * LDST + c] = f2bf(src[(long)(bg * 16 + r) * DIN + c]);
        }
    }
    __syncthreads();

    v8f c_st = {};                            // cell state (C-layout)

    // per-wave/lane fragment offsets (in ushorts) for the 4 gate tiles
    const long wo0 = (long)(0 * H16 + wv) * KT * 512 + lane * 16;
    const long wo1 = (long)(1 * H16 + wv) * KT * 512 + lane * 16;
    const long wo2 = (long)(2 * H16 + wv) * KT * 512 + lane * 16;
    const long wo3 = (long)(3 * H16 + wv) * KT * 512 + lane * 16;
    const unsigned short* arow = smem + m * LDST;

    for (int t = 0; t < T_; ++t) {
        // ---- stage current input tile [16 x DIN] into LDS (bf16) ----
        if (INMODE == IN_SEQ_F32) {
            const float* src = (const float*)in_seq;          // [B, T, DIN]
            for (int i = tid; i < 16 * DIN; i += nthreads) {
                int r = i / DIN, c = i % DIN;
                smem[r * LDST + c] =
                    f2bf(src[((long)(bg * 16 + r) * T_ + t) * DIN + c]);
            }
            if (t + 1 < T_ && tid < 16)
                __builtin_prefetch(src + ((long)(bg * 16 + tid) * T_ + (t + 1)) * DIN, 0, 0);
        } else if (INMODE == IN_SEQ_BF16) {
            const unsigned short* src = (const unsigned short*)in_seq;
            for (int i = tid; i < 16 * DIN; i += nthreads) {
                int r = i / DIN, c = i % DIN;
                smem[r * LDST + c] = src[((long)(bg * 16 + r) * T_ + t) * DIN + c];
            }
            if (t + 1 < T_ && tid < 16)
                __builtin_prefetch(src + ((long)(bg * 16 + tid) * T_ + (t + 1)) * DIN, 0, 0);
        }
        __syncthreads();

        // Opaque copy of the weight pointer: the empty asm must execute every
        // iteration and "may modify" wpi, so the compiler cannot hoist the
        // (t-invariant) weight-fragment loads out of the t-loop and spill
        // them to scratch. Loads stay normal (clause-able, L2-cached).
        unsigned long long wpi = (unsigned long long)wpack;
        asm volatile("" : "+s"(wpi));
        const unsigned short* wp = (const unsigned short*)wpi;
        const unsigned short* w0 = wp + wo0;
        const unsigned short* w1 = wp + wo1;
        const unsigned short* w2 = wp + wo2;
        const unsigned short* w3 = wp + wo3;

        // ---- gates = [x_t | h] @ Wcat^T via WMMA, K-loop over 32-wide tiles ----
        v8f ai = {}, af = {}, ag = {}, ao = {};
#pragma unroll
        for (int kt = 0; kt < KT; ++kt) {
            Frag32 fa;
            fa.lo = *(const uint4*)(arow + kt * 32 + hh * 16);
            fa.hi = *(const uint4*)(arow + kt * 32 + hh * 16 + 8);
            v16bf A = __builtin_bit_cast(v16bf, fa);

            Frag32 f0, f1, f2, f3;
            f0.lo = *(const uint4*)(w0 + kt * 512);
            f0.hi = *(const uint4*)(w0 + kt * 512 + 8);
            f1.lo = *(const uint4*)(w1 + kt * 512);
            f1.hi = *(const uint4*)(w1 + kt * 512 + 8);
            f2.lo = *(const uint4*)(w2 + kt * 512);
            f2.hi = *(const uint4*)(w2 + kt * 512 + 8);
            f3.lo = *(const uint4*)(w3 + kt * 512);
            f3.hi = *(const uint4*)(w3 + kt * 512 + 8);

            ai = __builtin_amdgcn_wmma_f32_16x16x32_bf16(
                false, A, false, __builtin_bit_cast(v16bf, f0), (short)0, ai, false, false);
            af = __builtin_amdgcn_wmma_f32_16x16x32_bf16(
                false, A, false, __builtin_bit_cast(v16bf, f1), (short)0, af, false, false);
            ag = __builtin_amdgcn_wmma_f32_16x16x32_bf16(
                false, A, false, __builtin_bit_cast(v16bf, f2), (short)0, ag, false, false);
            ao = __builtin_amdgcn_wmma_f32_16x16x32_bf16(
                false, A, false, __builtin_bit_cast(v16bf, f3), (short)0, ao, false, false);
        }
        __syncthreads();   // all A-fragment reads of old h done before overwrite

        // ---- activations + state update (C layout: m = v + 8*hh, n = lane&15) ----
        v8f hv;
#pragma unroll
        for (int v = 0; v < 8; ++v) {
            float I = fsigmoid(ai[v] + bi);
            float F = fsigmoid(af[v] + bf);
            float G = ftanh_fast(ag[v] + bg_);
            float O = fsigmoid(ao[v] + bo);
            float c = F * c_st[v] + I * G;
            c_st[v] = c;
            hv[v]   = O * ftanh_fast(c);
        }

        // ---- publish h to LDS (after the x columns) ----
#pragma unroll
        for (int v = 0; v < 8; ++v) {
            int row = v + 8 * hh;
            smem[row * LDST + DIN + col] = f2bf(hv[v]);
        }

        // ---- layer output ----
        if (OUTMODE == OUT_SEQ_BF16) {
            unsigned short* dst = (unsigned short*)out;       // [B, T, HID] bf16
#pragma unroll
            for (int v = 0; v < 8; ++v) {
                int row = v + 8 * hh;
                dst[((long)(bg * 16 + row) * T_ + t) * HID + col] = f2bf(hv[v]);
            }
        } else if (OUTMODE == OUT_SEQ_F32) {
            float* dst = (float*)out;                         // [B, T, HID] f32
#pragma unroll
            for (int v = 0; v < 8; ++v) {
                int row = v + 8 * hh;
                dst[((long)(bg * 16 + row) * T_ + t) * HID + col] = hv[v];
            }
        } else if (OUTMODE == OUT_LAST_F32) {
            if (t == T_ - 1) {
                float* dst = (float*)out;                     // [B, HID] f32
#pragma unroll
                for (int v = 0; v < 8; ++v) {
                    int row = v + 8 * hh;
                    dst[(long)(bg * 16 + row) * HID + col] = hv[v];
                }
            }
        }
        __syncthreads();
    }
}

// ---------------------------------------------------------------------------
// Host-side orchestration (graph-capture safe: only kernel launches on stream)
// ---------------------------------------------------------------------------
extern "C" void kernel_launch(void* const* d_in, const int* in_sizes, int n_in,
                              void* d_out, int out_size, void* d_ws, size_t ws_size,
                              hipStream_t stream) {
    (void)n_in; (void)out_size; (void)ws_size;

    const float* x   = (const float*)d_in[0];
    const float* eW0 = (const float*)d_in[1];
    const float* eU0 = (const float*)d_in[2];
    const float* eb0 = (const float*)d_in[3];
    const float* ec0 = (const float*)d_in[4];
    const float* eW1 = (const float*)d_in[5];
    const float* eU1 = (const float*)d_in[6];
    const float* eb1 = (const float*)d_in[7];
    const float* ec1 = (const float*)d_in[8];
    const float* dW0 = (const float*)d_in[9];
    const float* dU0 = (const float*)d_in[10];
    const float* db0 = (const float*)d_in[11];
    const float* dc0 = (const float*)d_in[12];
    const float* dW1 = (const float*)d_in[13];
    const float* dU1 = (const float*)d_in[14];
    const float* db1 = (const float*)d_in[15];
    const float* dc1 = (const float*)d_in[16];

    const int B = 128, DIN = 128, H = 256;
    const int T = in_sizes[0] / (B * DIN);   // 1024

    // workspace carve-up
    char* ws = (char*)d_ws;
    size_t off = 0;
    auto take = [&](size_t bytes) -> void* {
        void* p = ws + off;
        off = (off + bytes + 255) & ~(size_t)255;
        return p;
    };
    unsigned short* wp_e0 = (unsigned short*)take((size_t)4 * H   * (DIN + H)   * 2); // 1024x384
    unsigned short* wp_e1 = (unsigned short*)take((size_t)4 * H   * (H + H)     * 2); // 1024x512
    unsigned short* wp_d0 = (unsigned short*)take((size_t)4 * DIN * (H + DIN)   * 2); // 512x384
    unsigned short* wp_d1 = (unsigned short*)take((size_t)4 * DIN * (DIN + DIN) * 2); // 512x256
    float* bb_e0 = (float*)take((size_t)4 * H   * 4);
    float* bb_e1 = (float*)take((size_t)4 * H   * 4);
    float* bb_d0 = (float*)take((size_t)4 * DIN * 4);
    float* bb_d1 = (float*)take((size_t)4 * DIN * 4);
    unsigned short* y0 = (unsigned short*)take((size_t)B * T * H   * 2); // enc L0 out
    float*          hT = (float*)take((size_t)B * H * 4);               // enc L1 final h
    unsigned short* z0 = (unsigned short*)take((size_t)B * T * DIN * 2); // dec L0 out

    // pack weights + biases (cheap, once per launch)
    auto packw = [&](const float* Wih, const float* Whh, unsigned short* outw,
                     int din, int hid) {
        long total = (long)4 * hid * (din + hid);
        pack_weights<<<(int)((total + 255) / 256), 256, 0, stream>>>(Wih, Whh, outw, din, hid);
    };
    packw(eW0, eU0, wp_e0, DIN, H);
    packw(eW1, eU1, wp_e1, H,   H);
    packw(dW0, dU0, wp_d0, H,   DIN);
    packw(dW1, dU1, wp_d1, DIN, DIN);
    pack_bias<<<(4 * H   + 255) / 256, 256, 0, stream>>>(eb0, ec0, bb_e0, 4 * H);
    pack_bias<<<(4 * H   + 255) / 256, 256, 0, stream>>>(eb1, ec1, bb_e1, 4 * H);
    pack_bias<<<(4 * DIN + 255) / 256, 256, 0, stream>>>(db0, dc0, bb_d0, 4 * DIN);
    pack_bias<<<(4 * DIN + 255) / 256, 256, 0, stream>>>(db1, dc1, bb_d1, 4 * DIN);

    const int grid = B / 16;   // 8 persistent workgroups
    // encoder L0: x(f32) -> y0(bf16 sequence)
    lstm_layer<128, 256, IN_SEQ_F32, OUT_SEQ_BF16>
        <<<grid, 512, 0, stream>>>(x, wp_e0, bb_e0, y0, T);
    // encoder L1: y0 -> hT (final hidden only)
    lstm_layer<256, 256, IN_SEQ_BF16, OUT_LAST_F32>
        <<<grid, 512, 0, stream>>>(y0, wp_e1, bb_e1, hT, T);
    // decoder L0: broadcast hT -> z0(bf16 sequence)
    lstm_layer<256, 128, IN_BCAST_F32, OUT_SEQ_BF16>
        <<<grid, 256, 0, stream>>>(hT, wp_d0, bb_d0, z0, T);
    // decoder L1: z0 -> d_out (f32 sequence)
    lstm_layer<128, 128, IN_SEQ_BF16, OUT_SEQ_F32>
        <<<grid, 256, 0, stream>>>(z0, wp_d1, bb_d1, d_out, T);
}
